// cellula_12575664243143
// MI455X (gfx1250) — compile-verified, matrix operand order checked by reading, not run on hardware
//
#include <hip/hip_runtime.h>

typedef _Float16 f16;
typedef _Float16 h8  __attribute__((ext_vector_type(8)));
typedef _Float16 h16 __attribute__((ext_vector_type(16)));
typedef float    f8  __attribute__((ext_vector_type(8)));

#define TS 40   // LDS row stride in halves (80B rows: 16B-aligned, conflict-free bank rotation)
#define TR 18   // halo rows per 16x16 tile

// ---------------- reductions: min/max via order-preserving uint encoding ----------------
__device__ __forceinline__ unsigned encf(float f){
  unsigned u = __float_as_uint(f);
  return (u & 0x80000000u) ? ~u : (u | 0x80000000u);
}
__device__ __forceinline__ float decf(unsigned u){
  u = (u & 0x80000000u) ? (u & 0x7FFFFFFFu) : ~u;
  return __uint_as_float(u);
}

__global__ void k_init_scalars(unsigned* sc){
  sc[0] = 0xFFFFFFFFu;  // min accumulator (encoded)
  sc[1] = 0u;           // max accumulator (encoded)
}

__global__ __launch_bounds__(256) void k_minmax(const float* __restrict__ x, int n,
                                                unsigned* __restrict__ sc){
  __shared__ unsigned smn[256], smx[256];
  unsigned mn = 0xFFFFFFFFu, mx = 0u;
  for (int i = blockIdx.x*256 + threadIdx.x; i < n; i += gridDim.x*256){
    unsigned e = encf(x[i]);
    mn = e < mn ? e : mn;
    mx = e > mx ? e : mx;
  }
  int t = threadIdx.x;
  smn[t] = mn; smx[t] = mx;
  __syncthreads();
  for (int s = 128; s > 0; s >>= 1){
    if (t < s){
      if (smn[t+s] < smn[t]) smn[t] = smn[t+s];
      if (smx[t+s] > smx[t]) smx[t] = smx[t+s];
    }
    __syncthreads();
  }
  if (t == 0){ atomicMin(&sc[0], smn[0]); atomicMax(&sc[1], smx[0]); }
}

__global__ void k_fin_minmax(unsigned* __restrict__ scu, float* __restrict__ scf){
  float mn = decf(scu[0]), mx = decf(scu[1]);
  scf[2] = mn;
  scf[3] = 2.0f / (mx - mn);
}

__global__ __launch_bounds__(256) void k_norm(const float* __restrict__ x, f16* __restrict__ xn,
                                              const float* __restrict__ scf, int n){
  int i = blockIdx.x*256 + threadIdx.x;
  if (i < n) xn[i] = (f16)((x[i] - scf[2]) * scf[3] - 1.0f);
}

__global__ __launch_bounds__(256) void k_zero16(f16* __restrict__ p, int n){
  int i = blockIdx.x*256 + threadIdx.x;
  if (i < n) p[i] = (f16)0.0f;
}

// ---------------- 3x3 conv via WMMA Toeplitz: D = sum_dy A_dy x T_dy + C ----------------
// One wave per 16x16 output tile, 8 tiles per 256-thread block.
// LDS column c holds in(y0-1+r, x0-2+c) for c=0..19 (pairs 4B-aligned for async b32 gather);
// the Toeplitz band is shifted accordingly: T_dy[k,n] = w(dy, k-n-1) for k-n-1 in {0,1,2}.
// resample: 0 = direct, 1 = src is half-res (nearest upsample), 2 = src is 2x res (avg-pool 2)
__global__ __launch_bounds__(256) void k_conv3(const f16* __restrict__ src,
                                               f16* __restrict__ dst,
                                               const float* __restrict__ w9,
                                               const float* __restrict__ bias,
                                               const f16* __restrict__ bu,
                                               int W, int H, int resample, int doClip){
  __shared__ __align__(16) f16 lds[8 * TR * TS];
  const int lane = threadIdx.x & 31;
  const int wid  = threadIdx.x >> 5;
  const int tilesX = W >> 4;
  const int tile = blockIdx.x * 8 + wid;
  const int tx = tile % tilesX, ty = tile / tilesX;
  const int x0 = tx << 4, y0 = ty << 4;
  f16* tl = &lds[wid * (TR * TS)];

  const bool fast = (resample == 0) && (y0 >= 1) && (y0 + 17 <= H) &&
                    (x0 >= 2) && (x0 + 18 <= W);
  if (fast){
    // K slots 20..31 have zero Toeplitz rows but must still be non-NaN: zero-fill once.
    for (int i = lane; i < TR * 6; i += 32){
      int r = i / 6, q = i - r * 6;
      *(unsigned*)(&tl[r * TS + 20 + 2*q]) = 0u;
    }
    // Async gather of the 18x20 halo straight into LDS (ASYNCcnt path, no VGPR round trip).
    const unsigned ldsBase = (unsigned)(unsigned long long)(uintptr_t)(void*)tl;
    const f16* gb = src + (size_t)(y0 - 1) * W + (x0 - 2);
    for (int i = lane; i < TR * 10; i += 32){
      int r = i / 10, j = i - r * 10;
      unsigned la = ldsBase + (unsigned)(r * TS + 2*j) * 2u;
      unsigned long long ga = (unsigned long long)(uintptr_t)(gb + (size_t)r * W + 2*j);
      asm volatile("global_load_async_to_lds_b32 %0, %1, off" :: "v"(la), "v"(ga) : "memory");
    }
    asm volatile("s_wait_asynccnt 0x0" ::: "memory");
  } else {
    // Border / resampling fallback: predicated scalar staging with explicit zero padding.
    for (int i = lane; i < TR * 32; i += 32){
      int r = i >> 5, c = i & 31;
      float v = 0.0f;
      int gy = y0 - 1 + r, gx = x0 - 2 + c;
      if (c < 20 && (unsigned)gy < (unsigned)H && (unsigned)gx < (unsigned)W){
        if (resample == 0){
          v = (float)src[(size_t)gy * W + gx];
        } else if (resample == 1){
          v = (float)src[(size_t)(gy >> 1) * (W >> 1) + (gx >> 1)];
        } else {
          const int sw = W << 1;
          const f16* p = src + (size_t)(gy << 1) * sw + (gx << 1);
          v = 0.25f * ((float)p[0] + (float)p[1] + (float)p[sw] + (float)p[sw + 1]);
        }
      }
      tl[r * TS + c] = (f16)v;
    }
    asm volatile("" ::: "memory");  // same-wave DS ops are in-order; just pin compiler order
  }

  const int n  = lane & 15;   // output column within tile (B/C/D lane mapping)
  const int m  = lane & 15;   // A-row within tile
  const int hf = lane >> 4;   // lane-half selects K segments

  // Per-lane banded-Toeplitz B operands (band shifted by the x0-2 halo origin).
  f16 wh[9];
#pragma unroll
  for (int t = 0; t < 9; ++t) wh[t] = (f16)w9[t];
  h16 Bm[3];
#pragma unroll
  for (int e = 0; e < 16; ++e){
    int v = e >> 1;
    int k = ((v & 3) << 1) + (e & 1) + ((v >> 2) << 4) + (hf << 3);
    int dxx = k - n - 1;
#pragma unroll
    for (int dy = 0; dy < 3; ++dy){
      f16 wv = (dxx == 0) ? wh[dy*3+0] : (dxx == 1) ? wh[dy*3+1] : wh[dy*3+2];
      Bm[dy][e] = (dxx >= 0 && dxx <= 2) ? wv : (f16)0.0f;
    }
  }

  // C = bias + BU tile (f32 accumulator layout: lane gives N, VGPR r gives M=r+8*half)
  float cb = bias ? bias[0] : 0.0f;
  f8 acc;
#pragma unroll
  for (int r = 0; r < 8; ++r){
    int M = r + (hf << 3);
    float cv = cb;
    if (bu) cv += (float)bu[(size_t)(y0 + M) * W + (x0 + n)];
    acc[r] = cv;
  }

  // Three chained WMMAs: A_dy rows are halo rows m+dy; two aligned b128 LDS reads per A.
  const int seg = hf << 3;  // lanes 0-15: K[0..7]+K[16..23]; lanes 16-31: K[8..15]+K[24..31]
#pragma unroll
  for (int dy = 0; dy < 3; ++dy){
    const f16* row = &tl[(m + dy) * TS + seg];
    h8 lo  = *(const h8*)row;
    h8 hi2 = *(const h8*)(row + 16);
    h16 A = __builtin_shufflevector(lo, hi2, 0,1,2,3,4,5,6,7,8,9,10,11,12,13,14,15);
    acc = __builtin_amdgcn_wmma_f32_16x16x32_f16(false, A, false, Bm[dy],
                                                 (short)0, acc, false, false);
  }

#pragma unroll
  for (int r = 0; r < 8; ++r){
    int M = r + (hf << 3);
    float v = acc[r];
    if (doClip) v = fminf(fmaxf(v, -1.0f), 1.0f);
    dst[(size_t)(y0 + M) * W + (x0 + n)] = (f16)v;
  }
}

// ---------------- 1x1 cross-scale mixing (pointwise, resampling fused) ----------------
__global__ __launch_bounds__(256) void k_mix_x(const f16* __restrict__ xs, const f16* __restrict__ ds,
                                               const f16* __restrict__ us, const float* __restrict__ w,
                                               f16* __restrict__ o){
  int i = blockIdx.x*256 + threadIdx.x;      // 2048x2048
  int x = i & 2047, y = i >> 11;
  const f16* up = us + (size_t)(y << 1) * 4096 + (x << 1);
  float a = 0.25f * ((float)up[0] + (float)up[1] + (float)up[4096] + (float)up[4097]);
  float b = (float)xs[i];
  float c = (float)ds[(size_t)(y >> 1) * 1024 + (x >> 1)];
  o[i] = (f16)(w[0]*a + w[1]*b + w[2]*c);
}

__global__ __launch_bounds__(256) void k_mix_u(const f16* __restrict__ xs, const f16* __restrict__ ds,
                                               const f16* __restrict__ us, const float* __restrict__ w,
                                               f16* __restrict__ o){
  int i = blockIdx.x*256 + threadIdx.x;      // 4096x4096
  int x = i & 4095, y = i >> 12;
  float a = (float)us[i];
  float b = (float)xs[(size_t)(y >> 1) * 2048 + (x >> 1)];
  float c = (float)ds[(size_t)(y >> 2) * 1024 + (x >> 2)];
  o[i] = (f16)(w[0]*a + w[1]*b + w[2]*c);
}

__global__ __launch_bounds__(256) void k_mix_d(const f16* __restrict__ xs, const f16* __restrict__ ds,
                                               const f16* __restrict__ us, const float* __restrict__ w,
                                               f16* __restrict__ o){
  int i = blockIdx.x*256 + threadIdx.x;      // 1024x1024
  int x = i & 1023, y = i >> 10;
  const f16* up = us + (size_t)(y << 2) * 4096 + (x << 2);
  float a = 0.0f;
#pragma unroll
  for (int dy = 0; dy < 4; ++dy)
#pragma unroll
    for (int dx = 0; dx < 4; ++dx) a += (float)up[dy*4096 + dx];
  a *= 0.0625f;
  const f16* xp = xs + (size_t)(y << 1) * 2048 + (x << 1);
  float b = 0.25f * ((float)xp[0] + (float)xp[1] + (float)xp[2048] + (float)xp[2049]);
  float c = (float)ds[i];
  o[i] = (f16)(w[0]*a + w[1]*b + w[2]*c);
}

// ---------------- sums (deterministic fixed-block partials) ----------------
__global__ __launch_bounds__(256) void k_sum16(const f16* __restrict__ p, int n,
                                               float* __restrict__ part){
  __shared__ float sd[256];
  float s = 0.0f;
  for (int i = blockIdx.x*256 + threadIdx.x; i < n; i += gridDim.x*256) s += (float)p[i];
  sd[threadIdx.x] = s;
  __syncthreads();
  for (int st = 128; st > 0; st >>= 1){
    if (threadIdx.x < st) sd[threadIdx.x] += sd[threadIdx.x + st];
    __syncthreads();
  }
  if (threadIdx.x == 0) part[blockIdx.x] = sd[0];
}

__global__ void k_means(const float* __restrict__ part, float* __restrict__ scf){
  float su = 0.0f, sx = 0.0f, sd = 0.0f;
  for (int i = 0; i < 1024; ++i){ su += part[i]; sx += part[1024+i]; sd += part[2048+i]; }
  scf[4] = su / 16777216.0f;   // mean(u_dn) == mean(us)
  scf[5] = sx / 4194304.0f;    // mean(xs)
  scf[6] = sd / 1048576.0f;    // mean(d_up) == mean(ds)
}

// ---------------- final 1x1 output + threshold masks ----------------
__global__ __launch_bounds__(256) void k_final(const f16* __restrict__ xs, const f16* __restrict__ ds,
                                               const f16* __restrict__ us, const float* __restrict__ w3,
                                               const float* __restrict__ ob, const float* __restrict__ scf,
                                               float* __restrict__ out, int* __restrict__ um,
                                               int* __restrict__ xm, int* __restrict__ dm){
  int i = blockIdx.x*256 + threadIdx.x;      // 2048x2048
  int x = i & 2047, y = i >> 11;
  const f16* up = us + (size_t)(y << 1) * 4096 + (x << 1);
  float u_dn = 0.25f * ((float)up[0] + (float)up[1] + (float)up[4096] + (float)up[4097]);
  float xv = (float)xs[i];
  float dv = (float)ds[(size_t)(y >> 1) * 1024 + (x >> 1)];
  float o = w3[0]*u_dn + w3[1]*xv + w3[2]*dv + ob[0];
  out[i] = fminf(fmaxf(o, -1.0f), 1.0f);
  um[i] = u_dn > scf[4];
  xm[i] = xv   > scf[5];
  dm[i] = dv   > scf[6];
}

// ---------------- host orchestration ----------------
extern "C" void kernel_launch(void* const* d_in, const int* in_sizes, int n_in,
                              void* d_out, int out_size, void* d_ws, size_t ws_size,
                              hipStream_t stream){
  (void)in_sizes; (void)n_in; (void)out_size; (void)ws_size;
  const float* x       = (const float*)d_in[0];
  const float* convAx  = (const float*)d_in[1];
  const float* convBx  = (const float*)d_in[2];
  const float* convAu  = (const float*)d_in[3];
  const float* convBu  = (const float*)d_in[4];
  const float* convAd  = (const float*)d_in[5];
  const float* convBd  = (const float*)d_in[6];
  const float* biasx   = (const float*)d_in[7];
  const float* biasu   = (const float*)d_in[8];
  const float* biasd   = (const float*)d_in[9];
  const float* outBias = (const float*)d_in[10];
  const float* mixw[9];
  for (int i = 0; i < 9; ++i) mixw[i] = (const float*)d_in[11 + i]; // x5,x10,x15,u5,u10,u15,d5,d10,d15
  const float* conv1x1 = (const float*)d_in[20];

  const int HX = 2048, HD = 1024, HU = 4096;
  const int NX = HX*HX, ND = HD*HD, NU = HU*HU;

  char* ws = (char*)d_ws;
  unsigned* scu  = (unsigned*)ws;
  float*    scf  = (float*)ws;
  float*    part = (float*)(ws + 1024);          // 3*1024 partial sums
  size_t off = 65536;
  f16* xn  = (f16*)(ws + off); off += (size_t)NX * 2;
  f16* BUx = (f16*)(ws + off); off += (size_t)NX * 2;
  f16* BUd = (f16*)(ws + off); off += (size_t)ND * 2;
  f16* BUu = (f16*)(ws + off); off += (size_t)NU * 2;
  f16* xsb[2]; xsb[0] = (f16*)(ws + off); off += (size_t)NX * 2;
               xsb[1] = (f16*)(ws + off); off += (size_t)NX * 2;
  f16* dsb[2]; dsb[0] = (f16*)(ws + off); off += (size_t)ND * 2;
               dsb[1] = (f16*)(ws + off); off += (size_t)ND * 2;
  f16* usb[2]; usb[0] = (f16*)(ws + off); off += (size_t)NU * 2;
               usb[1] = (f16*)(ws + off); off += (size_t)NU * 2;

  dim3 blk(256);

  // normalize: xn = 2*(x-min)/(max-min) - 1
  k_init_scalars<<<1, 1, 0, stream>>>(scu);
  k_minmax<<<1024, blk, 0, stream>>>(x, NX, scu);
  k_fin_minmax<<<1, 1, 0, stream>>>(scu, scf);
  k_norm<<<NX/256, blk, 0, stream>>>(x, xn, scf, NX);

  // precomputed BU terms (resampling fused into conv halo staging)
  k_conv3<<<NX/2048, blk, 0, stream>>>(xn, BUx, convBx, nullptr, nullptr, HX, HX, 0, 0);
  k_conv3<<<ND/2048, blk, 0, stream>>>(xn, BUd, convBd, nullptr, nullptr, HD, HD, 2, 0);
  k_conv3<<<NU/2048, blk, 0, stream>>>(xn, BUu, convBu, nullptr, nullptr, HU, HU, 1, 0);

  // zero state
  k_zero16<<<NX/256, blk, 0, stream>>>(xsb[0], NX);
  k_zero16<<<ND/256, blk, 0, stream>>>(dsb[0], ND);
  k_zero16<<<NU/256, blk, 0, stream>>>(usb[0], NU);

  int ci = 0;
  for (int it = 0; it < 20; ++it){
    if (it == 5 || it == 10 || it == 15){
      int mi = (it == 5) ? 0 : (it == 10) ? 1 : 2;
      k_mix_x<<<NX/256, blk, 0, stream>>>(xsb[ci], dsb[ci], usb[ci], mixw[0+mi], xsb[ci^1]);
      k_mix_u<<<NU/256, blk, 0, stream>>>(xsb[ci], dsb[ci], usb[ci], mixw[3+mi], usb[ci^1]);
      k_mix_d<<<ND/256, blk, 0, stream>>>(xsb[ci], dsb[ci], usb[ci], mixw[6+mi], dsb[ci^1]);
      ci ^= 1;
    }
    k_conv3<<<NX/2048, blk, 0, stream>>>(xsb[ci], xsb[ci^1], convAx, biasx, BUx, HX, HX, 0, 1);
    k_conv3<<<ND/2048, blk, 0, stream>>>(dsb[ci], dsb[ci^1], convAd, biasd, BUd, HD, HD, 0, 1);
    k_conv3<<<NU/2048, blk, 0, stream>>>(usb[ci], usb[ci^1], convAu, biasu, BUu, HU, HU, 0, 1);
    ci ^= 1;
  }

  // means (deterministic two-stage reduction)
  k_sum16<<<1024, blk, 0, stream>>>(usb[ci], NU, part);
  k_sum16<<<1024, blk, 0, stream>>>(xsb[ci], NX, part + 1024);
  k_sum16<<<1024, blk, 0, stream>>>(dsb[ci], ND, part + 2048);
  k_means<<<1, 1, 0, stream>>>(part, scf);

  float* outF = (float*)d_out;
  int* um = (int*)d_out + (size_t)NX;
  int* xm = um + (size_t)NX;
  int* dm = xm + (size_t)NX;
  k_final<<<NX/256, blk, 0, stream>>>(xsb[ci], dsb[ci], usb[ci], conv1x1, outBias, scf,
                                      outF, um, xm, dm);
}
